// AttentionGate_22617297781349
// MI455X (gfx1250) — matrix operand order repaired and checked by reading, not compiled, not run on hardware
//
#include <hip/hip_runtime.h>

// ---- CDNA5 WMMA types -------------------------------------------------------
typedef __attribute__((ext_vector_type(16))) __bf16          v16bf;
typedef __attribute__((ext_vector_type(8)))  float           v8f;
typedef __attribute__((ext_vector_type(8)))  unsigned short  us8;

union ABf16 {             // A/B fragment: 16 bf16 = 8 VGPRs
    v16bf          bf;
    unsigned short u[16];
    us8            v8[2];
};

// float -> bf16, round-to-nearest-even (integer bit trick; the compiler pairs
// these into add3 + mov_b16 at 1.5 VALU/elem, which beat every alternative
// this toolchain can select)
__device__ __forceinline__ unsigned short f2bf(float f) {
    union { float f; unsigned u; } c; c.f = f;
    unsigned u = c.u;
    u += 0x7FFFu + ((u >> 16) & 1u);
    return (unsigned short)(u >> 16);
}

// hardware tanh (gfx1250 V_TANH_F32) when available, else libm
__device__ __forceinline__ float fast_tanh(float x) {
#if __has_builtin(__builtin_amdgcn_tanhf)
    return __builtin_amdgcn_tanhf(x);
#else
    return tanhf(x);
#endif
}

// Problem constants (B,T,D) = (128, 512, 256); K = 4*D = 1024; H = 50 (pad 64)
#define DDIM   256
#define TDIM   512
#define HVAL   50
#define HPAD   64
#define KBLK   256      // one z-block of K per LDS phase
#define KPAD   264      // LDS K padding (528B row stride -> conflict-free b128)
#define ROWS_PER_WG 64  // 4 waves x 16 rows

__global__ __launch_bounds__(128)
void attn_gate_fused(const float* __restrict__ facts,
                     const float* __restrict__ question,
                     const float* __restrict__ memory,
                     const float* __restrict__ W1,
                     const float* __restrict__ b1,
                     const float* __restrict__ W2,
                     const float* __restrict__ b2,
                     float* __restrict__ out)
{
    __shared__ unsigned short ldsW[HPAD * KPAD];   // W1^T tile, bf16, ~33 KB
    __shared__ float          ldsLogit[ROWS_PER_WG];

    const int tid  = threadIdx.x;
    const int lane = tid & 31;
    const int wv   = tid >> 5;        // wave -> 16-row sub-tile (0..3)
    const int mrow = lane & 15;       // A-row / B,C-column index within tile
    const int hi   = lane >> 4;       // lane half (selects K sub-ranges)

    const int b  = (int)blockIdx.x >> 3;          // batch (8 WGs per batch)
    const int t0 = ((int)blockIdx.x & 7) * ROWS_PER_WG;
    const int t  = t0 + (wv << 4) + mrow;         // this lane's fact row

    const float* frow = facts    + ((size_t)b * TDIM + t) * DDIM;
    const float* qrow = question + (size_t)b * DDIM;
    const float* mrw  = memory   + (size_t)b * DDIM;

    // one accumulator per N-tile; lane's column in tile j is n_j = 16*j + mrow
    v8f acc[4];
    #pragma unroll
    for (int j = 0; j < 4; ++j) {
        const int   nj  = (j << 4) + mrow;
        const float b1v = (nj < HVAL) ? b1[nj] : 0.0f;
        #pragma unroll
        for (int i = 0; i < 8; ++i) acc[j][i] = b1v;
    }

    if (tid < ROWS_PER_WG) ldsLogit[tid] = b2[0];

    // ---- 4 z-blocks: {f*q, f*m, |f-q|, |f-m|}, each K=256 -------------------
    for (int blk = 0; blk < 4; ++blk) {
        __syncthreads();                          // protect previous phase reads
        // Stage W1[blk*256 .. +255, 0..63] into LDS, transposed, bf16
        for (int idx = tid; idx < HPAD * KBLK; idx += 128) {
            const int nn = idx >> 8;
            const int kl = idx & 255;
            const float w = (nn < HVAL) ? W1[(size_t)(blk * KBLK + kl) * HVAL + nn]
                                        : 0.0f;
            ldsW[nn * KPAD + kl] = f2bf(w);
        }
        __syncthreads();

        const float* vrow  = (blk & 1) ? mrw : qrow;
        const bool   domul = (blk < 2);

        #pragma unroll 1
        for (int kc = 0; kc < 8; ++kc) {
            const int o0 = kc * 32 + (hi << 3);   // lane's first d in this chunk

            if (kc < 7)                           // pull next facts chunk early
                __builtin_prefetch(frow + o0 + 32, 0, 3);

            // facts + (q|m): two contiguous runs of 8 floats each
            float ff[16], vv[16];
            {
                const float4 f0 = *(const float4*)(frow + o0);
                const float4 f1 = *(const float4*)(frow + o0 + 4);
                const float4 f2 = *(const float4*)(frow + o0 + 16);
                const float4 f3 = *(const float4*)(frow + o0 + 20);
                const float4 q0 = *(const float4*)(vrow + o0);
                const float4 q1 = *(const float4*)(vrow + o0 + 4);
                const float4 q2 = *(const float4*)(vrow + o0 + 16);
                const float4 q3 = *(const float4*)(vrow + o0 + 20);
                ff[0]=f0.x; ff[1]=f0.y; ff[2]=f0.z; ff[3]=f0.w;
                ff[4]=f1.x; ff[5]=f1.y; ff[6]=f1.z; ff[7]=f1.w;
                ff[8]=f2.x; ff[9]=f2.y; ff[10]=f2.z; ff[11]=f2.w;
                ff[12]=f3.x; ff[13]=f3.y; ff[14]=f3.z; ff[15]=f3.w;
                vv[0]=q0.x; vv[1]=q0.y; vv[2]=q0.z; vv[3]=q0.w;
                vv[4]=q1.x; vv[5]=q1.y; vv[6]=q1.z; vv[7]=q1.w;
                vv[8]=q2.x; vv[9]=q2.y; vv[10]=q2.z; vv[11]=q2.w;
                vv[12]=q3.x; vv[13]=q3.y; vv[14]=q3.z; vv[15]=q3.w;
            }

            // A fragment (built ONCE, reused by 4 WMMAs): z packed to bf16
            ABf16 a;
            #pragma unroll
            for (int e = 0; e < 16; ++e) {
                const float z = domul ? ff[e] * vv[e] : fabsf(ff[e] - vv[e]);
                a.u[e] = f2bf(z);
            }

            // Load ALL 4 B fragments first (8 back-to-back ds_load_b128,
            // progressive dscnt waits), then 4 independent WMMAs.
            ABf16 bb[4];
            #pragma unroll
            for (int j = 0; j < 4; ++j) {
                const int nj = (j << 4) + mrow;
                const us8* wp =
                (const us8*)&ldsW[nj * KPAD + kc * 32 + (hi << 3)];
            bb[j].v8[0] = wp[0];
            bb[j].v8[1] = wp[2];
            }
            #pragma unroll
            for (int j = 0; j < 4; ++j) {
                acc[j] = __builtin_amdgcn_wmma_f32_16x16x32_bf16(
                             /*neg_a=*/false, a.bf, /*neg_b=*/false, bb[j].bf,
                             /*c_mod=*/(short)0, acc[j],
                             /*reuse_a=*/false, /*reuse_b=*/false);
            }
        }
    }

    // ---- tail: g = tanh(acc); logits = g @ W2 + b2; softmax(singleton) ------
    float w2v[4];
    #pragma unroll
    for (int j = 0; j < 4; ++j) {
        const int nj = (j << 4) + mrow;
        w2v[j] = (nj < HVAL) ? W2[nj] : 0.0f;
    }
    #pragma unroll
    for (int i = 0; i < 8; ++i) {
        float s = 0.0f;
        #pragma unroll
        for (int j = 0; j < 4; ++j) s += fast_tanh(acc[j][i]) * w2v[j];
        const int M = (wv << 4) + i + (hi << 3);   // row within WG's 64
        atomicAdd(&ldsLogit[M], s);                // ds_add_f32
    }
    __syncthreads();

    if (tid < ROWS_PER_WG) {
        const float l = ldsLogit[tid];
        const float e = __expf(l - l);             // softmax over axis of size 1
        out[(size_t)b * TDIM + t0 + tid] = e / e;  // == 1.0f
    }
}

extern "C" void kernel_launch(void* const* d_in, const int* in_sizes, int n_in,
                              void* d_out, int out_size, void* d_ws, size_t ws_size,
                              hipStream_t stream) {
    (void)in_sizes; (void)n_in; (void)d_ws; (void)ws_size; (void)out_size;
    const float* facts    = (const float*)d_in[0];
    const float* question = (const float*)d_in[1];
    const float* memory   = (const float*)d_in[2];
    const float* W1       = (const float*)d_in[3];
    const float* b1       = (const float*)d_in[4];
    const float* W2       = (const float*)d_in[5];
    const float* b2       = (const float*)d_in[6];
    float* out = (float*)d_out;

    const int nblocks = (128 * 512) / ROWS_PER_WG;   // 1024 workgroups
    attn_gate_fused<<<dim3(nblocks), dim3(128), 0, stream>>>(
        facts, question, memory, W1, b1, W2, b2, out);
}
